// Downsample_4286377361906
// MI455X (gfx1250) — compile-verified, matrix operand order checked by reading, not run on hardware
//
#include <hip/hip_runtime.h>
#include <hip/hip_bf16.h>
#include <cstdint>

// ---------------------------------------------------------------------------
// OctreeConv(2x2x2, stride2) + bias + BatchNorm1d(training) for MI455X gfx1250
//
// GEMM: M=65536 (nodes), K=2048 (8 children * 256 ch), N=512 (out ch)
// Strategy: bf16 WMMA (v_wmma_f32_16x16x32_bf16) with f32 accumulation.
//   - f32 inputs are converted to bf16 (RNE) while staging into LDS.
//   - block tile 128(M) x 64(N), K-step 32, 8 wave32 waves per block,
//     each wave computes a 32x32 tile as 2x2 WMMA accumulators.
//   - B is staged TRANSPOSED in LDS so both A and B fragments are two
//     16-byte ds_load_b128 per lane (per ISA 7.12.2 layouts).
//   - LDS row stride = 40 halves (80B): 20*lane mod 64 covers all 64 banks
//     exactly once across 16 lanes -> conflict-free fragment loads.
// BN is done deterministically (no float atomics): per-channel block
// reduction kernel, then an elementwise normalize kernel.
// ---------------------------------------------------------------------------

typedef __attribute__((ext_vector_type(16))) __bf16 v16bf;
typedef __attribute__((ext_vector_type(8)))  float  v8f;

#define CIN   256
#define COUT  512
#define KNB   8            // children per output node
#define KDIM  (KNB * CIN)  // 2048
#define BM    128
#define BN    64
#define BK    32
#define LDA   40           // padded half-stride for A rows in LDS
#define LDB   40           // padded half-stride for Bt rows in LDS
#define BN_EPS 1e-5f

// f32 -> bf16 with round-to-nearest-even (matches hardware conversion closely
// enough for data that immediately feeds a BatchNorm).
__device__ __forceinline__ unsigned short f2bf(float f) {
    unsigned u = __float_as_uint(f);
    unsigned r = u + 0x7FFFu + ((u >> 16) & 1u);
    return (unsigned short)(r >> 16);
}

union FragU {
    uint4 q[2];   // two 16B LDS loads (halves kb..kb+7 and kb+16..kb+23)
    v16bf v;
};

__global__ __launch_bounds__(256)
void octconv_gemm_kernel(const float* __restrict__ data,     // [N_in, CIN]
                         const float* __restrict__ weight,   // [KDIM, COUT]
                         const float* __restrict__ bias,     // [COUT]
                         const int*   __restrict__ neigh,    // [N_out, 8]
                         float* __restrict__ out)            // [N_out, COUT]
{
    __shared__ __align__(16) unsigned short lsA[BM * LDA];   // 10240 B
    __shared__ __align__(16) unsigned short lsB[BN * LDB];   // 5120 B (transposed: [n][k])

    const int tid  = threadIdx.x;
    const int lane = tid & 31;
    const int wave = tid >> 5;     // 0..7
    const int wm   = wave >> 1;    // 0..3 : 32-row group
    const int wn   = wave & 1;     // 0..1 : 32-col group
    const int bn0  = blockIdx.x * BN;
    const int bm0  = blockIdx.y * BM;

    const int lm = lane & 15;      // row (A) / col (B) within 16x16 tile
    const int hi = lane >> 4;      // half-wave selector
    const int kb = hi * 8;         // K-half base for fragment loads

    v8f acc[2][2] = {};            // 2x2 of 16x16 f32 accumulators

    for (int k0 = 0; k0 < KDIM; k0 += BK) {
        __syncthreads();           // protect LDS from previous iteration's readers

        // ---- stage A: gather 128 x 32 f32 via neigh, convert to bf16 ----
        // Within one K-tile (32 wide, 32-aligned) the child index is constant.
        {
            const int ksub  = k0 >> 8;     // which of the 8 children
            const int cbase = k0 & 255;    // channel offset inside that child
            #pragma unroll
            for (int i = 0; i < 4; ++i) {                 // 1024 float4 / 256 thr
                int eg = tid + i * 256;
                int m  = eg >> 3;                          // row in tile (32 cols = 8 float4)
                int c4 = eg & 7;
                int nidx = neigh[(bm0 + m) * KNB + ksub];
                const float4 v = *reinterpret_cast<const float4*>(
                    &data[(size_t)nidx * CIN + cbase + c4 * 4]);
                ushort4 h;
                h.x = f2bf(v.x); h.y = f2bf(v.y); h.z = f2bf(v.z); h.w = f2bf(v.w);
                *reinterpret_cast<ushort4*>(&lsA[m * LDA + c4 * 4]) = h;
            }
        }

        // ---- stage B: 32 x 64 f32 -> bf16, stored transposed Bt[n][k] ----
        #pragma unroll
        for (int i = 0; i < 2; ++i) {                     // 512 float4 / 256 thr
            int eg = tid + i * 256;
            int r  = eg >> 4;                              // k row (64 cols = 16 float4)
            int c4 = eg & 15;
            const float4 v = *reinterpret_cast<const float4*>(
                &weight[(size_t)(k0 + r) * COUT + bn0 + c4 * 4]);
            int n = c4 * 4;
            lsB[(n + 0) * LDB + r] = f2bf(v.x);
            lsB[(n + 1) * LDB + r] = f2bf(v.y);
            lsB[(n + 2) * LDB + r] = f2bf(v.z);
            lsB[(n + 3) * LDB + r] = f2bf(v.w);
        }
        __syncthreads();

        // ---- fragment loads (ISA 7.12.2 16-bit A 16x32 layout) + 4 WMMAs ----
        FragU aF[2], bF[2];
        #pragma unroll
        for (int i = 0; i < 2; ++i) {
            int m = wm * 32 + i * 16 + lm;
            aF[i].q[0] = *reinterpret_cast<const uint4*>(&lsA[m * LDA + kb]);
            aF[i].q[1] = *reinterpret_cast<const uint4*>(&lsA[m * LDA + kb + 16]);
        }
        #pragma unroll
        for (int j = 0; j < 2; ++j) {
            int n = wn * 32 + j * 16 + lm;
            bF[j].q[0] = *reinterpret_cast<const uint4*>(&lsB[n * LDB + kb]);
            bF[j].q[1] = *reinterpret_cast<const uint4*>(&lsB[n * LDB + kb + 16]);
        }
        #pragma unroll
        for (int i = 0; i < 2; ++i) {
            #pragma unroll
            for (int j = 0; j < 2; ++j) {
                acc[i][j] = __builtin_amdgcn_wmma_f32_16x16x32_bf16(
                    /*neg_a=*/false, aF[i].v, /*neg_b=*/false, bF[j].v,
                    /*c_mod=*/(short)0, acc[i][j],
                    /*reuse_a=*/false, /*reuse_b=*/false);
            }
        }
    }

    // ---- epilogue: add bias, store (C/D layout: VGPR r <-> row r + 8*hi) ----
    #pragma unroll
    for (int j = 0; j < 2; ++j) {
        const int gn = bn0 + wn * 32 + j * 16 + lm;
        const float bv = bias[gn];
        #pragma unroll
        for (int i = 0; i < 2; ++i) {
            const int rowBase = bm0 + wm * 32 + i * 16 + hi * 8;
            #pragma unroll
            for (int r = 0; r < 8; ++r) {
                out[(rowBase + r) * COUT + gn] = acc[i][j][r] + bv;
            }
        }
    }
}

// Deterministic per-channel batch statistics: one block per output channel,
// fixed-order serial accumulation per thread + fixed-pairing LDS tree.
__global__ __launch_bounds__(256)
void bn_stats_kernel(const float* __restrict__ out,
                     float* __restrict__ g_sum,   // [COUT]
                     float* __restrict__ g_sq,    // [COUT]
                     int n_rows)
{
    __shared__ float ssum[256];
    __shared__ float ssq[256];
    const int c = blockIdx.x;
    float s = 0.f, q = 0.f;
    for (int r = threadIdx.x; r < n_rows; r += 256) {
        float v = out[(size_t)r * COUT + c];
        s += v;
        q += v * v;
    }
    ssum[threadIdx.x] = s;
    ssq[threadIdx.x]  = q;
    __syncthreads();
    for (int off = 128; off > 0; off >>= 1) {
        if (threadIdx.x < off) {
            ssum[threadIdx.x] += ssum[threadIdx.x + off];
            ssq[threadIdx.x]  += ssq[threadIdx.x + off];
        }
        __syncthreads();
    }
    if (threadIdx.x == 0) {
        g_sum[c] = ssum[0];
        g_sq[c]  = ssq[0];
    }
}

// Elementwise BN apply: out = (out - mean) * rsqrt(var + eps) * gamma + beta
__global__ __launch_bounds__(256)
void bn_normalize_kernel(float* __restrict__ out,
                         const float* __restrict__ g_sum,
                         const float* __restrict__ g_sq,
                         const float* __restrict__ gamma,
                         const float* __restrict__ beta,
                         int n_rows)
{
    const float invN = 1.0f / (float)n_rows;
    const long long total = (long long)n_rows * COUT;
    const long long stride = (long long)gridDim.x * blockDim.x * 4;
    for (long long idx = ((long long)blockIdx.x * blockDim.x + threadIdx.x) * 4;
         idx < total; idx += stride) {
        const int c = (int)(idx & (COUT - 1));      // 4-aligned, COUT multiple of 4
        float4 v  = *reinterpret_cast<float4*>(&out[idx]);
        float4 sm = *reinterpret_cast<const float4*>(&g_sum[c]);
        float4 sq = *reinterpret_cast<const float4*>(&g_sq[c]);
        float4 gm = *reinterpret_cast<const float4*>(&gamma[c]);
        float4 bt = *reinterpret_cast<const float4*>(&beta[c]);

        float mx = sm.x * invN, my = sm.y * invN, mz = sm.z * invN, mw = sm.w * invN;
        float scx = rsqrtf(sq.x * invN - mx * mx + BN_EPS) * gm.x;
        float scy = rsqrtf(sq.y * invN - my * my + BN_EPS) * gm.y;
        float scz = rsqrtf(sq.z * invN - mz * mz + BN_EPS) * gm.z;
        float scw = rsqrtf(sq.w * invN - mw * mw + BN_EPS) * gm.w;

        v.x = (v.x - mx) * scx + bt.x;
        v.y = (v.y - my) * scy + bt.y;
        v.z = (v.z - mz) * scz + bt.z;
        v.w = (v.w - mw) * scw + bt.w;
        *reinterpret_cast<float4*>(&out[idx]) = v;
    }
}

extern "C" void kernel_launch(void* const* d_in, const int* in_sizes, int n_in,
                              void* d_out, int out_size, void* d_ws, size_t ws_size,
                              hipStream_t stream) {
    const float* data   = (const float*)d_in[0];
    const float* weight = (const float*)d_in[1];
    const float* bias   = (const float*)d_in[2];
    const float* gamma  = (const float*)d_in[3];
    const float* beta   = (const float*)d_in[4];
    const int*   neigh  = (const int*)d_in[5];   // JAX x64 off: int64 request -> int32
    float* out = (float*)d_out;

    float* g_sum = (float*)d_ws;            // [COUT]
    float* g_sq  = g_sum + COUT;            // [COUT]

    const int n_out = in_sizes[5] / KNB;    // 65536

    // 1) GEMM (+bias): grid x = N tiles (fast-varying so A rows stay hot in L2)
    dim3 grid(COUT / BN, n_out / BM);
    octconv_gemm_kernel<<<grid, 256, 0, stream>>>(data, weight, bias, neigh, out);

    // 2) deterministic per-channel stats
    bn_stats_kernel<<<COUT, 256, 0, stream>>>(out, g_sum, g_sq, n_out);

    // 3) normalize in place
    const long long total = (long long)n_out * COUT;
    int blocks = (int)((total / 4 + 255) / 256);
    bn_normalize_kernel<<<blocks, 256, 0, stream>>>(out, g_sum, g_sq, gamma, beta, n_out);
}
